// QLSTMGen_65481071396524
// MI455X (gfx1250) — matrix operand order repaired and checked by reading, not compiled, not run on hardware
//
#include <hip/hip_runtime.h>

// ---------------------------------------------------------------------------
// QLSTM on MI455X (gfx1250): split into
//   (1) data-parallel precompute  ZX = x @ Wx^T + b   (bf16 WMMA, whole GPU)
//   (2) recurrent kernel: 16 workgroups x 16 batch rows, weights resident in
//       LDS (bf16), h/c resident across all 512 steps, two WMMA phases per
//       step + fused gate math.
// ---------------------------------------------------------------------------

#define T_LEN 512
#define B_SZ  256
#define IN_D  128
#define HID_D 256
#define NQ_D  32
#define GN_D  128   // 4 * NQ
#define D_TOT 384

typedef __attribute__((ext_vector_type(16))) __bf16       v16bf;
typedef __attribute__((ext_vector_type(8)))  float        v8f;
typedef __attribute__((ext_vector_type(4)))  unsigned int u32x4;
typedef __attribute__((ext_vector_type(8)))  unsigned int u32x8;

static __device__ __forceinline__ unsigned short f2bf(float f) {
  unsigned int u = __builtin_bit_cast(unsigned int, f);
  u += 0x7FFFu + ((u >> 16) & 1u);           // round-to-nearest-even
  return (unsigned short)(u >> 16);
}

// A-matrix fragment (16x32 bf16): lane holds K = half*16 + {0..15}
// -> two contiguous 16B runs in an LDS row.
static __device__ __forceinline__ v16bf ld_a_frag(const unsigned short* p) {
  u32x4 lo = *(const u32x4*)p;
  u32x4 hi = *(const u32x4*)(p + 8);
  u32x8 w;
  w[0] = lo[0]; w[1] = lo[1]; w[2] = lo[2]; w[3] = lo[3];
  w[4] = hi[0]; w[5] = hi[1]; w[6] = hi[2]; w[7] = hi[3];
  return __builtin_bit_cast(v16bf, w);
}

// B-matrix fragment (32x16 bf16): lane holds K = half*16 + {0..15}, col = lane%16
// -> one contiguous 32B run in an LDS row (row-major over K).
static __device__ __forceinline__ v16bf ld_b_frag(const unsigned short* p) {
  return __builtin_bit_cast(v16bf, *(const u32x8*)p);
}

static __device__ __forceinline__ v8f wmma_bf16(v16bf a, v16bf b, v8f c) {
  return __builtin_amdgcn_wmma_f32_16x16x32_bf16(false, a, false, b, (short)0, c,
                                                 false, false);
}

static __device__ __forceinline__ float sigmoidf(float x) {
  return 1.0f / (1.0f + __expf(-x));
}

// ---------------------------------------------------------------------------
// Kernel A: ZX[t*B+b][n] = sum_k x[t][b][k] * Wg[n][k] + bg[n]   (k < 128)
// M = T*B = 131072, N = 128, K = 128.  One 128x128 output tile per workgroup.
// ---------------------------------------------------------------------------
__global__ __launch_bounds__(256) void zx_gemm_kernel(const float* __restrict__ X,
                                                      const float* __restrict__ Wg,
                                                      const float* __restrict__ bg,
                                                      float* __restrict__ ZX) {
  __shared__ unsigned short As[128][128];   // 32 KB  (M x K) bf16
  __shared__ unsigned short Ws[128][128];   // 32 KB  (N x K) bf16

  const int tid  = threadIdx.x;
  const int lane = tid & 31;
  const int wave = tid >> 5;
  const int half = lane >> 4;
  const int lm   = lane & 15;
  const long r0  = (long)blockIdx.x * 128;

  // Stage A tile (f32 -> bf16), float4 vectorized: 4096 float4 / 256 threads.
  for (int i = tid; i < 128 * 32; i += 256) {
    int row = i >> 5;
    int k4  = (i & 31) << 2;
    float4 v = *(const float4*)(X + (r0 + row) * IN_D + k4);
    unsigned short* d = &As[row][k4];
    d[0] = f2bf(v.x); d[1] = f2bf(v.y); d[2] = f2bf(v.z); d[3] = f2bf(v.w);
  }
  // Stage Wx tile: Wg rows are length 384; x-part is k in [0,128).
  for (int i = tid; i < 128 * 32; i += 256) {
    int n  = i >> 5;
    int k4 = (i & 31) << 2;
    float4 v = *(const float4*)(Wg + n * D_TOT + k4);
    unsigned short* d = &Ws[n][k4];
    d[0] = f2bf(v.x); d[1] = f2bf(v.y); d[2] = f2bf(v.z); d[3] = f2bf(v.w);
  }
  __syncthreads();

  v8f acc[8];
  for (int j = 0; j < 8; ++j)
    for (int v = 0; v < 8; ++v) acc[j][v] = 0.0f;

  const int mrow = wave * 16;  // this wave's 16 rows of the 128-row tile
  for (int kb = 0; kb < 4; ++kb) {
    v16bf a = ld_a_frag(&As[mrow + lm][kb * 32 + half * 16]);
    for (int j = 0; j < 8; ++j) {
      v16bf b = ld_b_frag(&Ws[j * 16 + lm][kb * 32 + half * 16]);
      acc[j] = wmma_bf16(a, b, acc[j]);
    }
  }

  // D layout: lane -> (row = v + 8*half, col = lane%16). Fold in bias.
  for (int j = 0; j < 8; ++j) {
    int n = j * 16 + lm;
    float bias = bg[n];
    for (int v = 0; v < 8; ++v) {
      long row = r0 + mrow + v + 8 * half;
      ZX[row * GN_D + n] = acc[j][v] + bias;
    }
  }
}

// ---------------------------------------------------------------------------
// Kernel B: recurrence. 16 workgroups x 16 batch rows x 512 steps.
// LDS: WhG (128x256 bf16, 64KB), Wh (256x32 bf16, 16KB), h (16x256 bf16, 8KB),
//      z scratch (16x128 bf16, 4KB), bh (1KB)  -> ~93KB of the 320KB WGP LDS.
// ---------------------------------------------------------------------------
__global__ __launch_bounds__(256) void qlstm_recurrent_kernel(
    const float* __restrict__ Wg, const float* __restrict__ Wh,
    const float* __restrict__ bh, const float* __restrict__ ZX,
    float* __restrict__ out) {
  __shared__ unsigned short WgH[128][256];  // (n, k) hidden part of W_gates
  __shared__ unsigned short WhS[256][32];   // (h, n) to_hidden projection
  __shared__ unsigned short hs[16][256];    // current h, bf16
  __shared__ unsigned short zs[16][128];    // per-step z, bf16
  __shared__ float bhS[256];

  const int tid  = threadIdx.x;
  const int lane = tid & 31;
  const int wave = tid >> 5;
  const int half = lane >> 4;
  const int lm   = lane & 15;
  const int b0   = blockIdx.x * 16;

  // --- one-time staging ---
  for (int i = tid; i < 128 * 64; i += 256) {   // WhG: k in [128,384) of Wg rows
    int n  = i >> 6;
    int k4 = (i & 63) << 2;
    float4 v = *(const float4*)(Wg + n * D_TOT + IN_D + k4);
    unsigned short* d = &WgH[n][k4];
    d[0] = f2bf(v.x); d[1] = f2bf(v.y); d[2] = f2bf(v.z); d[3] = f2bf(v.w);
  }
  for (int i = tid; i < 256 * 8; i += 256) {    // Wh (256 x 32)
    int h  = i >> 3;
    int k4 = (i & 7) << 2;
    float4 v = *(const float4*)(Wh + h * NQ_D + k4);
    unsigned short* d = &WhS[h][k4];
    d[0] = f2bf(v.x); d[1] = f2bf(v.y); d[2] = f2bf(v.z); d[3] = f2bf(v.w);
  }
  bhS[tid] = bh[tid];
  for (int i = tid; i < 16 * 256; i += 256) ((unsigned short*)hs)[i] = 0;

  v8f cc[2];                     // cell state: rows v+8*half, cols hcol+j*16+lm
  float hreg[2][8];
  for (int j = 0; j < 2; ++j)
    for (int v = 0; v < 8; ++v) { cc[j][v] = 0.0f; hreg[j][v] = 0.0f; }

  const int nz   = wave * 16;    // phase-1 column block (of 128)
  const int hcol = wave * 32;    // phase-2 hidden column base (of 256)
  __syncthreads();

  for (int t = 0; t < T_LEN; ++t) {
    // ---- phase 1: z = ZX(t) + h @ WhG^T  (16x128, K=256) ----
    const float* zxrow = ZX + ((long)t * B_SZ + b0) * GN_D;
    if (t + 1 < T_LEN) {  // hide next step's ZX fetch behind the WMMAs
      const float* nx = ZX + ((long)(t + 1) * B_SZ + b0) * GN_D;
      __builtin_prefetch(nx + (8 * half) * GN_D + nz + lm, 0, 3);
    }
    v8f acc;
    for (int v = 0; v < 8; ++v) acc[v] = zxrow[(v + 8 * half) * GN_D + nz + lm];

    for (int kb = 0; kb < 8; ++kb) {
      v16bf a = ld_a_frag(&hs[lm][kb * 32 + half * 16]);
      v16bf b = ld_b_frag(&WgH[nz + lm][kb * 32 + half * 16]);
      acc = wmma_bf16(a, b, acc);
    }
    for (int v = 0; v < 8; ++v)
      zs[v + 8 * half][nz + lm] = f2bf(acc[v]);
    __syncthreads();

    // ---- phase 2: per gate g, h4 = z[:, g*32:(g+1)*32] @ Wh^T + bh ----
    v8f g4[4][2];
    for (int g = 0; g < 4; ++g) {
      v16bf a = ld_a_frag(&zs[lm][g * 32 + half * 16]);
      for (int j = 0; j < 2; ++j) {
        float bb = bhS[hcol + j * 16 + lm];
        v8f c0;
        for (int v = 0; v < 8; ++v) c0[v] = bb;
        v16bf b = ld_b_frag(&WhS[hcol + j * 16 + lm][half * 16]);
        g4[g][j] = wmma_bf16(a, b, c0);
      }
    }

    // ---- fused gates + state update ----
    for (int j = 0; j < 2; ++j) {
      int col = hcol + j * 16 + lm;
      for (int v = 0; v < 8; ++v) {
        float f  = sigmoidf(g4[0][j][v]);
        float ig = sigmoidf(g4[1][j][v]);
        float gg = tanhf(g4[2][j][v]);
        float o  = sigmoidf(g4[3][j][v]);
        float cv = f * cc[j][v] + ig * gg;
        cc[j][v] = cv;
        float hv = o * tanhf(cv);
        hreg[j][v] = hv;
        int row = v + 8 * half;
        hs[row][col] = f2bf(hv);
        out[((long)t * B_SZ + b0 + row) * HID_D + col] = hv;
      }
    }
    __syncthreads();  // hs fully written before next step reads it
  }

  // final (hx, cx) appended after outputs
  float* hx = out + (long)T_LEN * B_SZ * HID_D;
  float* cx = hx + (long)B_SZ * HID_D;
  for (int j = 0; j < 2; ++j) {
    int col = hcol + j * 16 + lm;
    for (int v = 0; v < 8; ++v) {
      int row = v + 8 * half;
      hx[(b0 + row) * HID_D + col] = hreg[j][v];
      cx[(b0 + row) * HID_D + col] = cc[j][v];
    }
  }
}

extern "C" void kernel_launch(void* const* d_in, const int* in_sizes, int n_in,
                              void* d_out, int out_size, void* d_ws, size_t ws_size,
                              hipStream_t stream) {
  (void)in_sizes; (void)n_in; (void)out_size; (void)ws_size;
  const float* X  = (const float*)d_in[0];   // (512,256,128)
  const float* Wg = (const float*)d_in[1];   // (4,32,384) -> flat (128,384)
  const float* bg = (const float*)d_in[2];   // (128)
  const float* Wh = (const float*)d_in[3];   // (256,32)
  const float* bh = (const float*)d_in[4];   // (256)
  float* out = (float*)d_out;
  float* ZX  = (float*)d_ws;                 // 512*256*128 f32 = 64 MiB scratch

  zx_gemm_kernel<<<dim3((T_LEN * B_SZ) / 128), dim3(256), 0, stream>>>(X, Wg, bg, ZX);
  qlstm_recurrent_kernel<<<dim3(B_SZ / 16), dim3(256), 0, stream>>>(Wg, Wh, bh, ZX, out);
}
